// FlowAttention_84112639525331
// MI455X (gfx1250) — compile-verified, matrix operand order checked by reading, not run on hardware
//
#include <hip/hip_runtime.h>
#include <hip/hip_bf16.h>

#define BB 4
#define LLEN 4096
#define HH 8
#define DD 64
#define BH (BB*HH)
#define LSTRIDE (HH*DD)      // 512 floats between consecutive l for fixed (b,h)
#define FEPS 1e-6f
#define NCH1 16              // L-chunks for column-sum kernels
#define LCH (LLEN/NCH1)      // 256
#define SCH 8                // s-chunks for kv WMMA kernel
#define SCHL (LLEN/SCH)      // 512

typedef __attribute__((ext_vector_type(16))) _Float16 v16h;
typedef __attribute__((ext_vector_type(8)))  float    v8f;

__device__ __forceinline__ float sigf(float x) { return 1.0f / (1.0f + __expf(-x)); }

__device__ __forceinline__ float waveSum(float v) {
#pragma unroll
  for (int off = 16; off > 0; off >>= 1) v += __shfl_xor(v, off, 32);
  return v;
}

// ---------- Kernel 1a: partial column sums of sig(q), sig(k) over an L-chunk ----------
// grid = BH*NCH1 blocks x 256 threads
__global__ void fa_psums(const float* __restrict__ Q, const float* __restrict__ K,
                         float* __restrict__ pq, float* __restrict__ pk) {
  int bh = blockIdx.x >> 4;            // /NCH1
  int ch = blockIdx.x & (NCH1 - 1);
  int b = bh >> 3, h = bh & 7;
  int base = b * LLEN * LSTRIDE + h * DD;
  int d = threadIdx.x & 63;
  int g = threadIdx.x >> 6;            // 0..3
  int l0 = ch * LCH;
  float aq = 0.f, ak = 0.f;
  for (int l = l0 + g; l < l0 + LCH; l += 4) {
    int idx = base + l * LSTRIDE + d;
    aq += sigf(Q[idx]);
    ak += sigf(K[idx]);
  }
  __shared__ float sq[4][DD];
  __shared__ float sk[4][DD];
  sq[g][d] = aq; sk[g][d] = ak;
  __syncthreads();
  if (threadIdx.x < DD) {
    int t = threadIdx.x;
    int o = (bh * NCH1 + ch) * DD + t;
    pq[o] = sq[0][t] + sq[1][t] + sq[2][t] + sq[3][t];
    pk[o] = sk[0][t] + sk[1][t] + sk[2][t] + sk[3][t];
  }
}

// ---------- Kernel 1b/3b: reduce NCH1 partials for two arrays at once ----------
// grid = BH blocks x 64 threads
__global__ void fa_reduce_pairs(const float* __restrict__ pa, const float* __restrict__ pb,
                                float* __restrict__ da, float* __restrict__ db) {
  int bh = blockIdx.x;
  int t = threadIdx.x;
  float sa = 0.f, sb = 0.f;
#pragma unroll
  for (int ch = 0; ch < NCH1; ++ch) {
    sa += pa[(bh * NCH1 + ch) * DD + t];
    sb += pb[(bh * NCH1 + ch) * DD + t];
  }
  da[bh * DD + t] = sa;
  db[bh * DD + t] = sb;
}

// ---------- Kernel 2: norm_row = 1/((q+eps).(ksum+eps)), norm_col = 1/((k+eps).(qsum+eps))
// one wave per position p = (bh, l); lane covers d and d+32
__global__ void fa_norm1(const float* __restrict__ Q, const float* __restrict__ K,
                         const float* __restrict__ qsum, const float* __restrict__ ksum,
                         float* __restrict__ nrow, float* __restrict__ ncol) {
  int wave = threadIdx.x >> 5, lane = threadIdx.x & 31;
  int p = blockIdx.x * 8 + wave;       // 0 .. BH*LLEN-1
  int bh = p >> 12;
  int l = p & (LLEN - 1);
  int b = bh >> 3, h = bh & 7;
  int base = b * LLEN * LSTRIDE + h * DD + l * LSTRIDE;
  int d0 = lane, d1 = lane + 32;
  float ks0 = ksum[bh * DD + d0] + FEPS, ks1 = ksum[bh * DD + d1] + FEPS;
  float qs0 = qsum[bh * DD + d0] + FEPS, qs1 = qsum[bh * DD + d1] + FEPS;
  float rq = (sigf(Q[base + d0]) + FEPS) * ks0 + (sigf(Q[base + d1]) + FEPS) * ks1;
  float rk = (sigf(K[base + d0]) + FEPS) * qs0 + (sigf(K[base + d1]) + FEPS) * qs1;
  rq = waveSum(rq);
  rk = waveSum(rk);
  if (lane == 0) {
    nrow[p] = 1.0f / rq;
    ncol[p] = 1.0f / rk;
  }
}

// ---------- Kernel 3a: partial weighted column sums over an L-chunk ----------
// qr partial: sum_l sig(q)*nrow[l] ; kn partial: sum_s sig(k)*ncol[s]
__global__ void fa_pwsums(const float* __restrict__ Q, const float* __restrict__ K,
                          const float* __restrict__ nrow, const float* __restrict__ ncol,
                          float* __restrict__ pq, float* __restrict__ pk) {
  int bh = blockIdx.x >> 4;
  int ch = blockIdx.x & (NCH1 - 1);
  int b = bh >> 3, h = bh & 7;
  int base = b * LLEN * LSTRIDE + h * DD;
  int d = threadIdx.x & 63;
  int g = threadIdx.x >> 6;
  int l0 = ch * LCH;
  float aq = 0.f, ak = 0.f;
  for (int l = l0 + g; l < l0 + LCH; l += 4) {
    int idx = base + l * LSTRIDE + d;
    aq += sigf(Q[idx]) * nrow[bh * LLEN + l];
    ak += sigf(K[idx]) * ncol[bh * LLEN + l];
  }
  __shared__ float sq[4][DD];
  __shared__ float sk[4][DD];
  sq[g][d] = aq; sk[g][d] = ak;
  __syncthreads();
  if (threadIdx.x < DD) {
    int t = threadIdx.x;
    int o = (bh * NCH1 + ch) * DD + t;
    pq[o] = sq[0][t] + sq[1][t] + sq[2][t] + sq[3][t];
    pk[o] = sk[0][t] + sk[1][t] + sk[2][t] + sk[3][t];
  }
}

// ---------- Kernel 4: nrow_ref = sigmoid((q+eps).(knsum+eps)), ncol_raw = (k+eps).(qrsum+eps)
__global__ void fa_norm2(const float* __restrict__ Q, const float* __restrict__ K,
                         const float* __restrict__ qrsum, const float* __restrict__ knsum,
                         float* __restrict__ nrowref, float* __restrict__ ncolraw) {
  int wave = threadIdx.x >> 5, lane = threadIdx.x & 31;
  int p = blockIdx.x * 8 + wave;
  int bh = p >> 12;
  int l = p & (LLEN - 1);
  int b = bh >> 3, h = bh & 7;
  int base = b * LLEN * LSTRIDE + h * DD + l * LSTRIDE;
  int d0 = lane, d1 = lane + 32;
  float kn0 = knsum[bh * DD + d0] + FEPS, kn1 = knsum[bh * DD + d1] + FEPS;
  float qr0 = qrsum[bh * DD + d0] + FEPS, qr1 = qrsum[bh * DD + d1] + FEPS;
  float rq = (sigf(Q[base + d0]) + FEPS) * kn0 + (sigf(Q[base + d1]) + FEPS) * kn1;
  float rk = (sigf(K[base + d0]) + FEPS) * qr0 + (sigf(K[base + d1]) + FEPS) * qr1;
  rq = waveSum(rq);
  rk = waveSum(rk);
  if (lane == 0) {
    nrowref[p] = sigf(rq * ((float)LLEN / (float)LLEN)); // L/S == 1
    ncolraw[p] = rk;
  }
}

// ---------- Kernel 5: softmax stats (max, sumexp) over s per (b,h) ----------
__global__ void fa_smax_stats(const float* __restrict__ raw, float* __restrict__ stats) {
  int bh = blockIdx.x;
  __shared__ float red[256];
  float m = -1e30f;
  for (int s = threadIdx.x; s < LLEN; s += 256) m = fmaxf(m, raw[bh * LLEN + s]);
  red[threadIdx.x] = m;
  __syncthreads();
  for (int st = 128; st > 0; st >>= 1) {
    if (threadIdx.x < st) red[threadIdx.x] = fmaxf(red[threadIdx.x], red[threadIdx.x + st]);
    __syncthreads();
  }
  float mx = red[0];
  __syncthreads();
  float sm = 0.f;
  for (int s = threadIdx.x; s < LLEN; s += 256) sm += __expf(raw[bh * LLEN + s] - mx);
  red[threadIdx.x] = sm;
  __syncthreads();
  for (int st = 128; st > 0; st >>= 1) {
    if (threadIdx.x < st) red[threadIdx.x] += red[threadIdx.x + st];
    __syncthreads();
  }
  if (threadIdx.x == 0) { stats[bh * 2] = mx; stats[bh * 2 + 1] = red[0]; }
}

// ---------- Kernel 6: ncol_ref = softmax(raw) * S ----------
__global__ void fa_smax_apply(const float* __restrict__ raw, const float* __restrict__ stats,
                              float* __restrict__ ncolref) {
  int p = blockIdx.x * 256 + threadIdx.x;  // 0 .. BH*LLEN-1
  int bh = p >> 12;
  ncolref[p] = __expf(raw[p] - stats[bh * 2]) / stats[bh * 2 + 1] * (float)LLEN;
}

// ---------- Kernel 7a: partial kv over an s-chunk (WMMA) ----------
// grid = BH*SCH blocks x 512 threads = 16 waves; wave = one 16x16 tile of 64x64 kv.
__global__ void fa_kv_wmma(const float* __restrict__ K, const float* __restrict__ V,
                           const float* __restrict__ ncolref, float* __restrict__ pkv) {
  int bh = blockIdx.x >> 3;            // /SCH
  int sc = blockIdx.x & (SCH - 1);
  int b = bh >> 3, h = bh & 7;
  int base = b * LLEN * LSTRIDE + h * DD;
  int wave = threadIdx.x >> 5, lane = threadIdx.x & 31;
  int dt = wave >> 2, et = wave & 3;
  int mn = lane & 15, hi = lane >> 4;
  int d = dt * 16 + mn;   // A row (kv row)
  int e = et * 16 + mn;   // B col (kv col)
  int sbeg = sc * SCHL;
  v8f c = {};
  for (int s0 = sbeg; s0 < sbeg + SCHL; s0 += 32) {
    v16h a, bm;
#pragma unroll
    for (int j = 0; j < 8; ++j) {
      int kk = ((j & 4) << 2) + hi * 8 + ((j & 3) << 1);   // K index within the 32-chunk
      int s = s0 + kk;
      int ia = base + s * LSTRIDE + d;
      a[2 * j]     = (_Float16)sigf(K[ia]);
      a[2 * j + 1] = (_Float16)sigf(K[ia + LSTRIDE]);
      int ib = base + s * LSTRIDE + e;
      bm[2 * j]     = (_Float16)(V[ib] * ncolref[bh * LLEN + s]);
      bm[2 * j + 1] = (_Float16)(V[ib + LSTRIDE] * ncolref[bh * LLEN + s + 1]);
    }
    c = __builtin_amdgcn_wmma_f32_16x16x32_f16(false, a, false, bm, (short)0, c, false, false);
  }
#pragma unroll
  for (int j = 0; j < 8; ++j) {
    int mm = hi * 8 + j;                                   // C row within tile
    pkv[(bh * SCH + sc) * DD * DD + (dt * 16 + mm) * DD + (et * 16 + mn)] = c[j];
  }
}

// ---------- Kernel 7b: reduce SCH partial kv tiles ----------
// grid = BH blocks x 256 threads
__global__ void fa_kv_reduce(const float* __restrict__ pkv, float* __restrict__ kv) {
  int bh = blockIdx.x;
  for (int idx = threadIdx.x; idx < DD * DD; idx += 256) {
    float s = 0.f;
#pragma unroll
    for (int sc = 0; sc < SCH; ++sc) s += pkv[(bh * SCH + sc) * DD * DD + idx];
    kv[bh * DD * DD + idx] = s;
  }
}

// ---------- Kernel 8: x[l][e] = (sum_d sig(q[l,d])*kv[d,e]) * nrow[l] * nrowref[l] (WMMA)
// 32*256 blocks x 128 threads = 4 waves; block = one 16-row l-tile, wave = one 16-col e-tile.
__global__ void fa_out_wmma(const float* __restrict__ Q, const float* __restrict__ kv,
                            const float* __restrict__ nrow, const float* __restrict__ nrowref,
                            float* __restrict__ out) {
  int blk = blockIdx.x;
  int bh = blk >> 8;
  int lt = blk & 255;
  int b = bh >> 3, h = bh & 7;
  int base = b * LLEN * LSTRIDE + h * DD;
  int et = threadIdx.x >> 5, lane = threadIdx.x & 31;
  int mn = lane & 15, hi = lane >> 4;
  int l = lt * 16 + mn;
  v8f c = {};
#pragma unroll
  for (int k0 = 0; k0 < DD; k0 += 32) {
    v16h a, bm;
#pragma unroll
    for (int j = 0; j < 8; ++j) {
      int kk = k0 + ((j & 4) << 2) + hi * 8 + ((j & 3) << 1);  // d index
      int ia = base + l * LSTRIDE + kk;
      a[2 * j]     = (_Float16)sigf(Q[ia]);
      a[2 * j + 1] = (_Float16)sigf(Q[ia + 1]);
      bm[2 * j]     = (_Float16)kv[bh * DD * DD + kk * DD + et * 16 + mn];
      bm[2 * j + 1] = (_Float16)kv[bh * DD * DD + (kk + 1) * DD + et * 16 + mn];
    }
    c = __builtin_amdgcn_wmma_f32_16x16x32_f16(false, a, false, bm, (short)0, c, false, false);
  }
#pragma unroll
  for (int j = 0; j < 8; ++j) {
    int mm = hi * 8 + j;
    int lr = lt * 16 + mm;
    float sc = nrow[bh * LLEN + lr] * nrowref[bh * LLEN + lr];
    out[base + lr * LSTRIDE + et * 16 + mn] = c[j] * sc;
  }
}

extern "C" void kernel_launch(void* const* d_in, const int* in_sizes, int n_in,
                              void* d_out, int out_size, void* d_ws, size_t ws_size,
                              hipStream_t stream) {
  (void)in_sizes; (void)n_in; (void)out_size; (void)ws_size;
  const float* Q = (const float*)d_in[0];
  const float* K = (const float*)d_in[1];
  const float* V = (const float*)d_in[2];
  float* out = (float*)d_out;

  float* ws = (float*)d_ws;
  float* pq      = ws;                         // BH*NCH1*DD = 32768
  float* pk      = pq + BH * NCH1 * DD;        // 32768
  float* qsum    = pk + BH * NCH1 * DD;        // 2048
  float* ksum    = qsum + BH * DD;             // 2048
  float* qrsum   = ksum + BH * DD;             // 2048
  float* knsum   = qrsum + BH * DD;            // 2048
  float* stats   = knsum + BH * DD;            // 64
  float* nrow    = stats + 128;                // 131072
  float* ncol    = nrow + BH * LLEN;           // 131072
  float* nrowref = ncol + BH * LLEN;           // 131072
  float* ncolraw = nrowref + BH * LLEN;        // 131072
  float* ncolref = ncolraw + BH * LLEN;        // 131072
  float* pkv     = ncolref + BH * LLEN;        // BH*SCH*DD*DD = 1048576
  float* kv      = pkv + BH * SCH * DD * DD;   // 131072

  fa_psums       <<<BH * NCH1,       256, 0, stream>>>(Q, K, pq, pk);
  fa_reduce_pairs<<<BH,               64, 0, stream>>>(pq, pk, qsum, ksum);
  fa_norm1       <<<BH * LLEN / 8,   256, 0, stream>>>(Q, K, qsum, ksum, nrow, ncol);
  fa_pwsums      <<<BH * NCH1,       256, 0, stream>>>(Q, K, nrow, ncol, pq, pk);
  fa_reduce_pairs<<<BH,               64, 0, stream>>>(pq, pk, qrsum, knsum);
  fa_norm2       <<<BH * LLEN / 8,   256, 0, stream>>>(Q, K, qrsum, knsum, nrowref, ncolraw);
  fa_smax_stats  <<<BH,              256, 0, stream>>>(ncolraw, stats);
  fa_smax_apply  <<<BH * LLEN / 256, 256, 0, stream>>>(ncolraw, stats, ncolref);
  fa_kv_wmma     <<<BH * SCH,        512, 0, stream>>>(K, V, ncolref, pkv);
  fa_kv_reduce   <<<BH,              256, 0, stream>>>(pkv, kv);
  fa_out_wmma    <<<BH * 256,        128, 0, stream>>>(Q, kv, nrow, nrowref, out);
}
